// NonLocalBlock_15779709846206
// MI455X (gfx1250) — compile-verified, matrix operand order checked by reading, not run on hardware
//
#include <hip/hip_runtime.h>
#include <hip/hip_bf16.h>

// ---------------------------------------------------------------------------
// Non-local block (self-attention) for MI455X / gfx1250, wave32 + WMMA f16.
//   B=4, C=256, H=W=64 (N=4096), INTER=128
// Pipeline:
//   cvt_w_kernel : weights f32 -> f16 (one-time, removes cvt from GEMM loops)
//   xpose_kernel : x [B,C,N] f32 -> xT [B,N,C] f16 (LDS-tiled transpose)
//   proj_kernel  : theta[N,I], phiT[N,I], gT[I,N] f16 via v_wmma_f32_16x16x32_f16
//   attn_kernel  : flash-style softmax(theta phi^T) g -> y[N,I] f16
//                  (never materializes the 256MB attention matrix);
//                  B operands software-pipelined around the softmax VALU work
//   out_kernel   : x + w_out @ y -> out f32 (vectorized b128 loads/stores)
// ---------------------------------------------------------------------------

typedef _Float16 v16h __attribute__((ext_vector_type(16)));
typedef _Float16 v8h  __attribute__((ext_vector_type(8)));
typedef float    v8f  __attribute__((ext_vector_type(8)));

constexpr int NB = 4;      // batch
constexpr int NC = 256;    // channels
constexpr int NN = 4096;   // H*W
constexpr int NI = 128;    // inter channels

__device__ __forceinline__ v8f wmma16(v16h a, v16h b, v8f c) {
  // D(f32 16x16) = A(f16 16x32) * B(f16 32x16) + C
  return __builtin_amdgcn_wmma_f32_16x16x32_f16(
      /*neg_a=*/false, a, /*neg_b=*/false, b,
      /*c_mod=*/(short)0, c, /*reuse_a=*/false, /*reuse_b=*/false);
}

// Load a v16h operand from contiguous f16 memory following the CDNA5 WMMA
// 16-bit operand layout: halves 0..7 at p[0..7], halves 8..15 at p[16..23].
// (Caller folds the per-lane-half K offset of 0 or 8 into p.)
__device__ __forceinline__ v16h load_h16(const _Float16* p) {
  v8h lo = *(const v8h*)(p);
  v8h hi = *(const v8h*)(p + 16);
  return __builtin_shufflevector(lo, hi, 0, 1, 2, 3, 4, 5, 6, 7,
                                 8, 9, 10, 11, 12, 13, 14, 15);
}

// ---------------------------------------------------------------------------
// One-time weight conversion f32 -> f16 (all four weights are I*C = 32768).
// ---------------------------------------------------------------------------
__global__ void __launch_bounds__(256)
cvt_w_kernel(const float* __restrict__ a, const float* __restrict__ b,
             const float* __restrict__ c, const float* __restrict__ d,
             _Float16* __restrict__ oa, _Float16* __restrict__ ob,
             _Float16* __restrict__ oc, _Float16* __restrict__ od) {
  const int i = blockIdx.x * 256 + threadIdx.x;   // 0 .. 32767
  oa[i] = (_Float16)a[i];
  ob[i] = (_Float16)b[i];
  oc[i] = (_Float16)c[i];
  od[i] = (_Float16)d[i];
}

// ---------------------------------------------------------------------------
// Transpose + convert: x [B,C,N] f32 -> xT [B,N,C] f16, 32x32 LDS tiles.
// ---------------------------------------------------------------------------
__global__ void __launch_bounds__(256)
xpose_kernel(const float* __restrict__ x, _Float16* __restrict__ xT) {
  __shared__ _Float16 tile[32][33];               // pad to dodge bank conflicts
  const int ntn = NN / 32;                        // 128
  const int ntc = NC / 32;                        // 8
  const int b = blockIdx.x / (ntn * ntc);
  const int rem = blockIdx.x % (ntn * ntc);
  const int c0 = (rem / ntn) * 32;
  const int n0 = (rem % ntn) * 32;
  const int tx = threadIdx.x & 31;
  const int ty = threadIdx.x >> 5;                // 0..7

  const float* xb = x + (size_t)b * NC * NN;
#pragma unroll
  for (int s = 0; s < 4; ++s)                     // coalesced along n
    tile[ty + s * 8][tx] = (_Float16)xb[(size_t)(c0 + ty + s * 8) * NN + n0 + tx];
  __syncthreads();
  _Float16* xtb = xT + (size_t)b * NN * NC;
#pragma unroll
  for (int s = 0; s < 4; ++s)                     // coalesced along c
    xtb[(size_t)(n0 + ty + s * 8) * NC + c0 + tx] = tile[tx][ty + s * 8];
}

// ---------------------------------------------------------------------------
// Projection: out[n,i] = sum_c w[i,c] * x[c,n]   (one 16-row n-tile per wave)
// All operands f16; B tiles preloaded as a batch so the 8 WMMAs run
// back-to-back behind one load clause instead of waiting per-step.
// ---------------------------------------------------------------------------
__device__ __forceinline__ void proj_one(const v16h (&Ax)[8],
                                         const _Float16* __restrict__ wp,
                                         int lane, int nBase,
                                         _Float16* __restrict__ outp,
                                         bool transposed) {
  const int col = lane & 15;
  const int hh = (lane >> 4) << 3;          // K-half offset: 0 or 8
  const int rbase = (lane < 16) ? 0 : 8;    // D rows 0-7 vs 8-15
  for (int ic = 0; ic < 8; ++ic) {          // 8 chunks of 16 output channels
    const int i = ic * 16 + col;
    const _Float16* wrow = wp + (size_t)i * NC + hh;
    v16h Bw[8];
#pragma unroll
    for (int k = 0; k < 8; ++k) Bw[k] = load_h16(wrow + k * 32);
    v8f acc = {};
#pragma unroll
    for (int k = 0; k < 8; ++k) acc = wmma16(Ax[k], Bw[k], acc);
    if (!transposed) {                      // [N,I]: per-row scatter (u16)
#pragma unroll
      for (int r = 0; r < 8; ++r)
        outp[(size_t)(nBase + r + rbase) * NI + i] = (_Float16)acc[r];
    } else {                                // [I,N]: rows contiguous -> b128
      v8h hv;
#pragma unroll
      for (int r = 0; r < 8; ++r) hv[r] = (_Float16)acc[r];
      *(v8h*)(outp + (size_t)i * NN + nBase + rbase) = hv;
    }
  }
}

__global__ void __launch_bounds__(128)
proj_kernel(const _Float16* __restrict__ xT,
            const _Float16* __restrict__ wth, const _Float16* __restrict__ wph,
            const _Float16* __restrict__ wgh,
            _Float16* __restrict__ th, _Float16* __restrict__ pt,
            _Float16* __restrict__ gt) {
  const int lane = threadIdx.x & 31;
  const int wv = threadIdx.x >> 5;
  const int tile = blockIdx.x * 4 + wv;            // 1024 tiles total
  const int b = tile / (NN / 16);
  const int nBase = (tile % (NN / 16)) * 16;
  const int col = lane & 15;
  const int hh = (lane >> 4) << 3;

  // A operand: xT rows [16 n x 256 c], contiguous f16 b128 loads.
  const _Float16* xRow = xT + ((size_t)b * NN + nBase + col) * NC + hh;
  v16h Ax[8];
#pragma unroll
  for (int k = 0; k < 8; ++k) Ax[k] = load_h16(xRow + k * 32);

  proj_one(Ax, wth, lane, nBase, th + (size_t)b * NN * NI, false);
  proj_one(Ax, wph, lane, nBase, pt + (size_t)b * NN * NI, false);
  proj_one(Ax, wgh, lane, nBase, gt + (size_t)b * NI * NN, true);
}

// ---------------------------------------------------------------------------
// Flash attention: per wave, 16 query rows vs all 4096 keys (steps of 32).
// Online softmax; P transposed D-layout -> A-layout via per-wave LDS tile.
// phi^T B-tiles pipelined one key-block ahead; g B-tiles issued under the
// softmax VALU work so both WMMA chains run without load stalls.
// ---------------------------------------------------------------------------
__global__ void __launch_bounds__(128)
attn_kernel(const _Float16* __restrict__ th, const _Float16* __restrict__ pt,
            const _Float16* __restrict__ gt, _Float16* __restrict__ y) {
  __shared__ __align__(16) _Float16 Pbuf[4][16 * 32];  // 1KB per wave

  const int lane = threadIdx.x & 31;
  const int wv = threadIdx.x >> 5;
  const int tile = blockIdx.x * 4 + wv;
  const int b = tile / (NN / 16);
  const int qBase = (tile % (NN / 16)) * 16;
  const int col = lane & 15;
  const int hh = (lane >> 4) << 3;
  const int rbase = (lane < 16) ? 0 : 8;

  // Preload theta rows for this query tile: A operand, K = I = 128.
  const _Float16* thRow = th + ((size_t)b * NN + qBase + col) * NI + hh;
  v16h At[4];
#pragma unroll
  for (int k = 0; k < 4; ++k) At[k] = load_h16(thRow + k * 32);

  v8f O[8];
#pragma unroll
  for (int ci = 0; ci < 8; ++ci) O[ci] = v8f{};
  float m8[8], l8[8];
#pragma unroll
  for (int r = 0; r < 8; ++r) { m8[r] = -1e30f; l8[r] = 0.f; }

  const _Float16* ptB = pt + (size_t)b * NN * NI;
  const _Float16* gtB = gt + (size_t)b * NI * NN;

  // Prologue: phi^T tiles for key block 0.
  v16h Bp[8];
  {
    const _Float16* r0 = ptB + (size_t)col * NI + hh;
    const _Float16* r1 = ptB + (size_t)(16 + col) * NI + hh;
#pragma unroll
    for (int k = 0; k < 4; ++k) {
      Bp[k] = load_h16(r0 + k * 32);
      Bp[4 + k] = load_h16(r1 + k * 32);
    }
  }

  for (int kb = 0; kb < NN; kb += 32) {
    // S = theta @ phi^T for 32 keys (two 16-wide key sub-tiles),
    // operands already resident from the previous iteration.
    v8f S0 = {}, S1 = {};
#pragma unroll
    for (int k = 0; k < 4; ++k) {
      S0 = wmma16(At[k], Bp[k], S0);
      S1 = wmma16(At[k], Bp[4 + k], S1);
    }

    // Issue this block's g tiles now: their latency hides under the softmax.
    v16h Bg[8];
#pragma unroll
    for (int ci = 0; ci < 8; ++ci) {
      const _Float16* grow = gtB + (size_t)(ci * 16 + col) * NN + kb + hh;
      __builtin_prefetch(grow + 64, 0, 1);  // 2 key-blocks ahead on same row
      Bg[ci] = load_h16(grow);
    }

    __syncthreads();  // previous iteration's P reads done before we store
#pragma unroll
    for (int r = 0; r < 8; ++r) {
      // Row (query) stats: reduce across the 16 lanes of each half
      // (rows 0-7 live in lanes 0-15, rows 8-15 in lanes 16-31 — matches O).
      float mx = fmaxf(S0[r], S1[r]);
      mx = fmaxf(mx, __shfl_xor(mx, 1));
      mx = fmaxf(mx, __shfl_xor(mx, 2));
      mx = fmaxf(mx, __shfl_xor(mx, 4));
      mx = fmaxf(mx, __shfl_xor(mx, 8));
      const float mn = fmaxf(m8[r], mx);
      const float sc = __expf(m8[r] - mn);
      const float p0 = __expf(S0[r] - mn);
      const float p1 = __expf(S1[r] - mn);
      float ps = p0 + p1;
      ps += __shfl_xor(ps, 1);
      ps += __shfl_xor(ps, 2);
      ps += __shfl_xor(ps, 4);
      ps += __shfl_xor(ps, 8);
      l8[r] = l8[r] * sc + ps;
      m8[r] = mn;
#pragma unroll
      for (int ci = 0; ci < 8; ++ci) O[ci][r] *= sc;  // rescale accumulators
      const int row = r + rbase;
      Pbuf[wv][row * 32 + col] = (_Float16)p0;
      Pbuf[wv][row * 32 + 16 + col] = (_Float16)p1;
    }
    __syncthreads();

    // P (16x32) back in A-operand layout from LDS
    const v16h AP = load_h16(&Pbuf[wv][col * 32 + hh]);

    // Refill phi^T tiles for the next key block (pipelined ahead).
    if (kb + 32 < NN) {
      const _Float16* n0 = ptB + (size_t)(kb + 32 + col) * NI + hh;
      const _Float16* n1 = ptB + (size_t)(kb + 48 + col) * NI + hh;
#pragma unroll
      for (int k = 0; k < 4; ++k) {
        Bp[k] = load_h16(n0 + k * 32);
        Bp[4 + k] = load_h16(n1 + k * 32);
      }
    }

    // O += P @ g  (g stored transposed [I,N] so keys are contiguous)
#pragma unroll
    for (int ci = 0; ci < 8; ++ci) O[ci] = wmma16(AP, Bg[ci], O[ci]);
  }

  // Normalize and emit y[n, i] (f16)
#pragma unroll
  for (int ci = 0; ci < 8; ++ci) {
#pragma unroll
    for (int r = 0; r < 8; ++r) {
      const int row = r + rbase;
      y[((size_t)b * NN + qBase + row) * NI + ci * 16 + col] =
          (_Float16)(O[ci][r] / l8[r]);
    }
  }
}

// ---------------------------------------------------------------------------
// Output projection + residual: out[c,n] = x[c,n] + sum_i w_out[c,i] y[n,i]
// D rows are contiguous n -> vectorized b128 residual loads and stores.
// ---------------------------------------------------------------------------
__global__ void __launch_bounds__(128)
out_kernel(const _Float16* __restrict__ y, const _Float16* __restrict__ wouth,
           const float* __restrict__ x, float* __restrict__ out) {
  const int lane = threadIdx.x & 31;
  const int wv = threadIdx.x >> 5;
  const int tile = blockIdx.x * 4 + wv;
  const int b = tile / (NN / 16);
  const int nBase = (tile % (NN / 16)) * 16;
  const int col = lane & 15;
  const int hh = (lane >> 4) << 3;
  const int rbase = (lane < 16) ? 0 : 8;

  const _Float16* yRow = y + ((size_t)b * NN + nBase + col) * NI + hh;
  v16h Ay[4];
#pragma unroll
  for (int k = 0; k < 4; ++k) Ay[k] = load_h16(yRow + k * 32);

  for (int cc = 0; cc < 16; ++cc) {   // 256 output channels in tiles of 16
    const int c = cc * 16 + col;
    const size_t base = ((size_t)b * NC + c) * NN + nBase + rbase;
    // Preload B tiles and the residual so the WMMA chain runs stall-free.
    v16h Bw[4];
#pragma unroll
    for (int k = 0; k < 4; ++k)
      Bw[k] = load_h16(wouth + (size_t)c * NI + k * 32 + hh);
    const float4 x0 = *(const float4*)(x + base);
    const float4 x1 = *(const float4*)(x + base + 4);
    v8f acc = {};
#pragma unroll
    for (int k = 0; k < 4; ++k) acc = wmma16(Ay[k], Bw[k], acc);
    float4 o0, o1;
    o0.x = x0.x + acc[0]; o0.y = x0.y + acc[1];
    o0.z = x0.z + acc[2]; o0.w = x0.w + acc[3];
    o1.x = x1.x + acc[4]; o1.y = x1.y + acc[5];
    o1.z = x1.z + acc[6]; o1.w = x1.w + acc[7];
    *(float4*)(out + base) = o0;
    *(float4*)(out + base + 4) = o1;
  }
}

// ---------------------------------------------------------------------------
extern "C" void kernel_launch(void* const* d_in, const int* in_sizes, int n_in,
                              void* d_out, int out_size, void* d_ws,
                              size_t ws_size, hipStream_t stream) {
  (void)in_sizes; (void)n_in; (void)out_size; (void)ws_size;
  const float* x       = (const float*)d_in[0];
  const float* w_theta = (const float*)d_in[1];
  const float* w_phi   = (const float*)d_in[2];
  const float* w_g     = (const float*)d_in[3];
  const float* w_out   = (const float*)d_in[4];
  float* out = (float*)d_out;

  // Workspace layout (f16):
  //   weights 4 x 32768 (64KB each=256KB) | xT 8MB | th 4MB | pt 4MB |
  //   gT 4MB | y 4MB  => ~24.3 MB total
  _Float16* wth = (_Float16*)d_ws;
  _Float16* wph = wth + (size_t)NI * NC;
  _Float16* wgh = wph + (size_t)NI * NC;
  _Float16* wou = wgh + (size_t)NI * NC;
  _Float16* xT  = wou + (size_t)NC * NI;           // [B,N,C]
  _Float16* th  = xT + (size_t)NB * NN * NC;       // theta  [B,N,I]
  _Float16* pt  = th + (size_t)NB * NN * NI;       // phi^T  [B,N,I]
  _Float16* gt  = pt + (size_t)NB * NN * NI;       // g^T    [B,I,N]
  _Float16* yb  = gt + (size_t)NB * NN * NI;       // y      [B,N,I]

  cvt_w_kernel<<<(NI * NC) / 256, 256, 0, stream>>>(
      w_theta, w_phi, w_g, w_out, wth, wph, wgh, wou);
  xpose_kernel<<<NB * (NC / 32) * (NN / 32), 256, 0, stream>>>(x, xT);

  const int blocks = (NB * (NN / 16)) / 4;         // 256 blocks, 4 waves each
  proj_kernel<<<blocks, 128, 0, stream>>>(xT, wth, wph, wgh, th, pt, gt);
  attn_kernel<<<blocks, 128, 0, stream>>>(th, pt, gt, yb);
  out_kernel<<<blocks, 128, 0, stream>>>(yb, wou, x, out);
}
